// semanticpro_32873679683972
// MI455X (gfx1250) — matrix operand order repaired and checked by reading, not compiled, not run on hardware
//
#include <hip/hip_runtime.h>

typedef unsigned short u16;
typedef unsigned int   u32;
typedef __attribute__((ext_vector_type(16))) __bf16 v16bf;
typedef __attribute__((ext_vector_type(8)))  float  v8f;

union Frag { u32 u[8]; v16bf v; };

#define BB   8
#define CC   1024
#define NN   4096
#define KK   1024
#define NBG  3072
#define DCC  256
#define DI   256
#define NHH  4
#define HDIM 64
#define MLPD 512

// gfx1250 async global->LDS path (ASYNCcnt-tracked), guarded so the file
// still builds on toolchains that don't declare the builtins.
#if __has_builtin(__builtin_amdgcn_global_load_async_to_lds_b32) && \
    __has_builtin(__builtin_amdgcn_s_wait_asynccnt)
#define HAVE_ASYNC_LDS 1
typedef __attribute__((address_space(1))) int* gptr_t;
typedef __attribute__((address_space(3))) int* lptr_t;
#else
#define HAVE_ASYNC_LDS 0
#endif

__device__ __forceinline__ u16 f2bf(float f) {
  u32 u = __float_as_uint(f);
  u32 r = u + 0x7fffu + ((u >> 16) & 1u);
  return (u16)(r >> 16);
}

__device__ __forceinline__ float geluf(float x) {
  return 0.5f * x * (1.0f + erff(x * 0.70710678118654752f));
}

__device__ __forceinline__ v8f wmma_bf16(const Frag& a, const Frag& b, v8f c) {
  return __builtin_amdgcn_wmma_f32_16x16x32_bf16(false, a.v, false, b.v,
                                                 (short)0, c, false, false);
}

// ---------------------------------------------------------------------------
// Index build: stable partition == argsort(!mask) split at K
// ---------------------------------------------------------------------------
__global__ void build_indices_k(const unsigned char* __restrict__ mask,
                                int* __restrict__ d_idx, int* __restrict__ g_idx) {
  int b = blockIdx.x;
  if (threadIdx.x == 0) {
    int di = 0, gi = 0;
    for (int n = 0; n < NN; ++n) {
      if (mask[b * NN + n]) d_idx[b * KK + di++] = n;
      else                  g_idx[b * NBG + gi++] = n;
    }
  }
}

__global__ void gather_defect_k(const float* __restrict__ feat, const int* __restrict__ d_idx,
                                float* __restrict__ x1, float* __restrict__ x2) {
  size_t t = (size_t)blockIdx.x * 256 + threadIdx.x;
  int b = (int)(t / ((size_t)KK * CC));
  int rem = (int)(t % ((size_t)KK * CC));
  int tok = rem / CC, c = rem % CC;
  int src = d_idx[b * KK + tok];
  float v = feat[((size_t)b * NN + src) * CC + c];
  x1[t] = v; x2[t] = v;
}

__global__ void gather_bg_k(const float* __restrict__ feat, const int* __restrict__ g_idx,
                            u16* __restrict__ bg) {
  size_t t = (size_t)blockIdx.x * 256 + threadIdx.x;
  int b = (int)(t / ((size_t)NBG * CC));
  int rem = (int)(t % ((size_t)NBG * CC));
  int tok = rem / CC, c = rem % CC;
  int src = g_idx[b * NBG + tok];
  bg[t] = f2bf(feat[((size_t)b * NN + src) * CC + c]);
}

// bilinear (align_corners) sample of 32x32 depth at the defect positions only
__global__ void depth_gather_k(const float* __restrict__ depth, const int* __restrict__ d_idx,
                               u16* __restrict__ out) {
  int b = blockIdx.x / KK;
  int t = blockIdx.x % KK;
  int c = threadIdx.x; // 0..255
  int p = d_idx[b * KK + t];
  int h = p >> 6, w = p & 63;
  float fy = h * (31.0f / 63.0f);
  float fx = w * (31.0f / 63.0f);
  int y0 = (int)floorf(fy); int y1 = min(y0 + 1, 31); float wy = fy - y0;
  int x0 = (int)floorf(fx); int x1 = min(x0 + 1, 31); float wx = fx - x0;
  const float* dch = depth + (((size_t)b * DCC + c) * 32) * 32;
  float v00 = dch[y0 * 32 + x0], v01 = dch[y0 * 32 + x1];
  float v10 = dch[y1 * 32 + x0], v11 = dch[y1 * 32 + x1];
  float top = v00 * (1.0f - wx) + v01 * wx;
  float bot = v10 * (1.0f - wx) + v11 * wx;
  out[(size_t)blockIdx.x * DCC + c] = f2bf(top * (1.0f - wy) + bot * wy);
}

__global__ void f32_to_bf16_k(const float* __restrict__ in, u16* __restrict__ out) {
  size_t t = (size_t)blockIdx.x * 256 + threadIdx.x;
  out[t] = f2bf(in[t]);
}

// conv_w (C=1024, DC=256) -> bf16 transposed (DC, C)
__global__ void transpose_cw_k(const float* __restrict__ in, u16* __restrict__ out) {
  size_t t = (size_t)blockIdx.x * 256 + threadIdx.x;
  int c = (int)(t / CC);   // 0..255
  int o = (int)(t % CC);   // 0..1023
  out[t] = f2bf(in[(size_t)o * DCC + c]);
}

// ---------------------------------------------------------------------------
// LayerNorm: one row (C=1024) per block, f32 in -> bf16 out
// ---------------------------------------------------------------------------
__global__ void layernorm_k(const float* __restrict__ x, const float* __restrict__ g,
                            const float* __restrict__ be, u16* __restrict__ out) {
  __shared__ float s1[256], s2[256];
  size_t row = blockIdx.x;
  const float* xr = x + row * CC;
  float a = 0.f, b2 = 0.f;
  for (int c = threadIdx.x; c < CC; c += 256) { float v = xr[c]; a += v; b2 += v * v; }
  s1[threadIdx.x] = a; s2[threadIdx.x] = b2;
  __syncthreads();
  for (int st = 128; st > 0; st >>= 1) {
    if (threadIdx.x < st) { s1[threadIdx.x] += s1[threadIdx.x + st]; s2[threadIdx.x] += s2[threadIdx.x + st]; }
    __syncthreads();
  }
  float mean = s1[0] * (1.0f / CC);
  float var  = s2[0] * (1.0f / CC) - mean * mean;
  float rstd = rsqrtf(var + 1e-5f);
  for (int c = threadIdx.x; c < CC; c += 256)
    out[row * CC + c] = f2bf((xr[c] - mean) * rstd * g[c] + be[c]);
}

// ---------------------------------------------------------------------------
// bf16 WMMA GEMM: out = act(A(MxK) @ W(KxN) + bias)
// block = 256 threads (8 waves), tile 128x64, BK=64.
// waves in 4x2 grid, each wave owns a 32x32 tile -> 8 WMMAs per staging round.
// ---------------------------------------------------------------------------
__global__ void gemm_bf16_k(const u16* __restrict__ A, const u16* __restrict__ W,
                            const float* __restrict__ bias,
                            float* __restrict__ outF, u16* __restrict__ outH,
                            int M, int Kd, int Nn, int act, int addto) {
  __shared__ u16 As[128][66];
  __shared__ u16 Bt[64][66]; // transposed: Bt[n][k]

  const int tid  = threadIdx.x;
  const int lane = tid & 31;
  const int wave = tid >> 5;
  const int lm   = lane & 15;
  const int half = lane >> 4;
  const int ka   = half * 8;   // A fragment k-half offset
  const int kb16 = half * 16;  // unused; B fragment uses ka as well
  const int mb = blockIdx.y * 128;
  const int nb = blockIdx.x * 64;
  const int m0 = (wave >> 1) * 32;
  const int n0 = (wave & 1) * 32;

  v8f zero = {0.f, 0.f, 0.f, 0.f, 0.f, 0.f, 0.f, 0.f};
  v8f acc00 = zero, acc01 = zero, acc10 = zero, acc11 = zero;

  for (int kb = 0; kb < Kd; kb += 64) {
    __syncthreads();
    // ---- stage A tile 128x64 (4096 dwords, 16 per thread) ----
#pragma unroll
    for (int i = 0; i < 16; ++i) {
      int e = i * 256 + tid;       // 0..4095
      int row = e >> 5;            // 0..127
      int cp  = e & 31;            // 0..31 k-pair
      const u16* src = A + (size_t)(mb + row) * Kd + kb + cp * 2;
      if (i == 0) __builtin_prefetch(src + 64, 0, 1); // next K-slab (speculative)
#if HAVE_ASYNC_LDS
      __builtin_amdgcn_global_load_async_to_lds_b32(
          (gptr_t)(u16*)src, (lptr_t)&As[row][cp * 2], 0, 0);
#else
      *(u32*)&As[row][cp * 2] = *(const u32*)src;
#endif
    }
    // ---- stage B tile 64x64 transposed (2048 dwords, 8 per thread) ----
#pragma unroll
    for (int i = 0; i < 8; ++i) {
      int e = i * 256 + tid;
      int k  = e >> 5;             // 0..63
      int np = e & 31;             // 0..31 n-pair
      const u16* src = W + (size_t)(kb + k) * Nn + nb + np * 2;
      if (i == 0) __builtin_prefetch(src + (size_t)64 * Nn, 0, 1);
      u32 d = *(const u32*)src;
      Bt[np * 2][k]     = (u16)(d & 0xffffu);
      Bt[np * 2 + 1][k] = (u16)(d >> 16);
    }
#if HAVE_ASYNC_LDS
    __builtin_amdgcn_s_wait_asynccnt(0);
#endif
    __syncthreads();

#pragma unroll
    for (int ks = 0; ks < 2; ++ks) {
      Frag a0, a1, b0, b1;
#pragma unroll
      for (int j = 0; j < 8; ++j) {
        int kkA = (j >> 2) * 16 + (j & 3) * 2 + ka + ks * 32;
        a0.u[j] = *(const u32*)&As[m0 + lm][kkA];
        a1.u[j] = *(const u32*)&As[m0 + 16 + lm][kkA];
        int kkB = (j >> 2) * 16 + (j & 3) * 2 + ka + ks * 32;
        b0.u[j] = *(const u32*)&Bt[n0 + lm][kkB];
        b1.u[j] = *(const u32*)&Bt[n0 + 16 + lm][kkB];
      }
      acc00 = wmma_bf16(a0, b0, acc00);
      acc01 = wmma_bf16(a0, b1, acc01);
      acc10 = wmma_bf16(a1, b0, acc10);
      acc11 = wmma_bf16(a1, b1, acc11);
    }
  }

#pragma unroll
  for (int mi = 0; mi < 2; ++mi) {
#pragma unroll
    for (int ni = 0; ni < 2; ++ni) {
      v8f acc = mi ? (ni ? acc11 : acc10) : (ni ? acc01 : acc00);
      int coln = nb + n0 + ni * 16 + lm;
      float bval = bias ? bias[coln] : 0.f;
#pragma unroll
      for (int r = 0; r < 8; ++r) {
        int rowm = mb + m0 + mi * 16 + r + 8 * half;
        float v = acc[r] + bval;
        if (act == 1) v = geluf(v);
        size_t oidx = (size_t)rowm * Nn + coln;
        if (outF) { if (addto) outF[oidx] += v; else outF[oidx] = v; }
        if (outH) outH[oidx] = f2bf(v);
      }
    }
  }
}

// ---------------------------------------------------------------------------
// Flash attention, head_dim=64. Q/K/V/O are bf16 (rows, 256) with row =
// b*Ntok + tok, col = head*64 + d. Block = 4 waves; each wave owns 16 queries.
// ---------------------------------------------------------------------------
__global__ void flash_attn_k(const u16* __restrict__ Q, const u16* __restrict__ Kb,
                             const u16* __restrict__ Vb, u16* __restrict__ O,
                             int NQ, int NK, float scale) {
  __shared__ u16 Vt[64][34];       // transposed V chunk: Vt[hd][key]
  __shared__ u16 Pw[4][16][34];    // per-wave P tile: [q][key]

  const int nqt = NQ / 64;
  int qt = blockIdx.x % nqt;
  int h  = (blockIdx.x / nqt) % NHH;
  int b  = blockIdx.x / (nqt * NHH);

  const int lane = threadIdx.x & 31;
  const int wave = threadIdx.x >> 5;
  const int lm   = lane & 15;
  const int half = lane >> 4;
  const int ka   = half * 8;
  const int kb16 = half * 16;
  const int qbase = qt * 64 + wave * 16;

  // Q fragments (d 0..31 and 32..63), row = qbase + lm
  Frag qa0, qa1;
  const u16* Qrow = Q + ((size_t)b * NQ + qbase + lm) * 256 + h * HDIM;
#pragma unroll
  for (int j = 0; j < 8; ++j) {
    int kkA = (j >> 2) * 16 + (j & 3) * 2 + ka;
    qa0.u[j] = *(const u32*)(Qrow + kkA);
    qa1.u[j] = *(const u32*)(Qrow + 32 + kkA);
  }

  v8f o0 = {0.f,0.f,0.f,0.f,0.f,0.f,0.f,0.f};
  v8f o1 = o0, o2 = o0, o3 = o0;
  float mrow[8], lrow[8];
#pragma unroll
  for (int r = 0; r < 8; ++r) { mrow[r] = -3.0e38f; lrow[r] = 0.f; }

  for (int kc = 0; kc < NK; kc += 32) {
    __syncthreads();
    // stage V chunk transposed into LDS (cooperative, 128 threads)
    for (int e = threadIdx.x; e < 32 * 32; e += 128) {
      int krow = e >> 5;          // key 0..31
      int np   = e & 31;          // hd pair 0..31
      const u16* vsrc = Vb + ((size_t)b * NK + kc + krow) * 256 + h * HDIM + np * 2;
      u32 d = *(const u32*)vsrc;
      Vt[np * 2][krow]     = (u16)(d & 0xffffu);
      Vt[np * 2 + 1][krow] = (u16)(d >> 16);
    }
    __syncthreads();

    // S = Q @ K^T for this 32-key chunk (two 16x16 fragments)
    v8f s0 = {0.f,0.f,0.f,0.f,0.f,0.f,0.f,0.f};
    v8f s1 = s0;
#pragma unroll
    for (int s = 0; s < 2; ++s) {
      const u16* Krow = Kb + ((size_t)b * NK + kc + s * 16 + lm) * 256 + h * HDIM;
      Frag kb0, kb1;
#pragma unroll
      for (int j = 0; j < 8; ++j) {
        int kkB = (j >> 2) * 16 + (j & 3) * 2 + ka;
        kb0.u[j] = *(const u32*)(Krow + kkB);
        kb1.u[j] = *(const u32*)(Krow + 32 + kkB);
      }
      if (s == 0) { s0 = wmma_bf16(qa0, kb0, s0); s0 = wmma_bf16(qa1, kb1, s0); }
      else        { s1 = wmma_bf16(qa0, kb0, s1); s1 = wmma_bf16(qa1, kb1, s1); }
    }

    // online softmax update (row stats shared across each 16-lane half)
#pragma unroll
    for (int r = 0; r < 8; ++r) {
      float a = s0[r] * scale, c = s1[r] * scale;
      float v = fmaxf(a, c);
#pragma unroll
      for (int off = 8; off >= 1; off >>= 1) v = fmaxf(v, __shfl_xor(v, off, 16));
      float mnew = fmaxf(mrow[r], v);
      float corr = __expf(mrow[r] - mnew);
      float e0 = __expf(a - mnew);
      float e1 = __expf(c - mnew);
      float rs = e0 + e1;
#pragma unroll
      for (int off = 8; off >= 1; off >>= 1) rs += __shfl_xor(rs, off, 16);
      lrow[r] = lrow[r] * corr + rs;
      mrow[r] = mnew;
      o0[r] *= corr; o1[r] *= corr; o2[r] *= corr; o3[r] *= corr;
      // stage P (D-layout -> LDS [q][key])
      Pw[wave][r + 8 * half][lm]      = f2bf(e0);
      Pw[wave][r + 8 * half][16 + lm] = f2bf(e1);
    }

    // P fragment (A-layout) from per-wave LDS region (same-wave DS in order)
    Frag pa;
#pragma unroll
    for (int j = 0; j < 8; ++j) {
      int kkA = (j >> 2) * 16 + (j & 3) * 2 + ka;
      pa.u[j] = *(const u32*)&Pw[wave][lm][kkA];
    }
    // V fragments (B-layout) from transposed LDS tile
#pragma unroll
    for (int f = 0; f < 4; ++f) {
      Frag vb;
#pragma unroll
      for (int j = 0; j < 8; ++j) {
        int kkB = (j >> 2) * 16 + (j & 3) * 2 + ka;
        vb.u[j] = *(const u32*)&Vt[f * 16 + lm][kkB];
      }
      if (f == 0) o0 = wmma_bf16(pa, vb, o0);
      else if (f == 1) o1 = wmma_bf16(pa, vb, o1);
      else if (f == 2) o2 = wmma_bf16(pa, vb, o2);
      else o3 = wmma_bf16(pa, vb, o3);
    }
  }

  // normalize + store O (bf16)
#pragma unroll
  for (int f = 0; f < 4; ++f) {
    v8f oo = (f == 0) ? o0 : (f == 1) ? o1 : (f == 2) ? o2 : o3;
    int coln = h * HDIM + f * 16 + lm;
#pragma unroll
    for (int r = 0; r < 8; ++r) {
      int rowm = qbase + r + 8 * half;
      O[((size_t)b * NQ + rowm) * 256 + coln] = f2bf(oo[r] / lrow[r]);
    }
  }
}

// ---------------------------------------------------------------------------
// Final scatter: out[b, d_idx, :] = x1 + 0.4 * x2
// ---------------------------------------------------------------------------
__global__ void scatter_out_k(const float* __restrict__ x1, const float* __restrict__ x2,
                              const int* __restrict__ d_idx, float* __restrict__ out) {
  size_t t = (size_t)blockIdx.x * 256 + threadIdx.x;
  int b = (int)(t / ((size_t)KK * CC));
  int rem = (int)(t % ((size_t)KK * CC));
  int tok = rem / CC, c = rem % CC;
  int pos = d_idx[b * KK + tok];
  out[((size_t)b * NN + pos) * CC + c] = x1[t] + 0.4f * x2[t];
}

// ---------------------------------------------------------------------------
extern "C" void kernel_launch(void* const* d_in, const int* in_sizes, int n_in,
                              void* d_out, int out_size, void* d_ws, size_t ws_size,
                              hipStream_t stream) {
  (void)in_sizes; (void)n_in; (void)ws_size;
  const float* feat  = (const float*)d_in[0];
  const unsigned char* mask = (const unsigned char*)d_in[1];
  const float* depth = (const float*)d_in[2];
  auto F = [&](int i) { return (const float*)d_in[i]; };

  char* wsb = (char*)d_ws;
  size_t off = 0;
  auto alloc = [&](size_t bytes) -> void* {
    void* p = wsb + off; off = (off + bytes + 255) & ~(size_t)255; return p;
  };

  int*   d_idx = (int*)alloc(sizeof(int) * BB * KK);
  int*   g_idx = (int*)alloc(sizeof(int) * BB * NBG);
  float* x1    = (float*)alloc(sizeof(float) * (size_t)BB * KK * CC);
  float* x2    = (float*)alloc(sizeof(float) * (size_t)BB * KK * CC);
  u16*   xln   = (u16*)alloc(2ull * BB * KK * CC);
  u16*   bg    = (u16*)alloc(2ull * BB * NBG * CC);
  u16*   ddin  = (u16*)alloc(2ull * BB * KK * DCC);
  u16*   ddef  = (u16*)alloc(2ull * BB * KK * CC);
  u16*   qb    = (u16*)alloc(2ull * BB * KK * DI);
  u16*   kbuf  = (u16*)alloc(2ull * BB * KK * DI);
  u16*   vbuf  = (u16*)alloc(2ull * BB * KK * DI);
  u16*   kc    = (u16*)alloc(2ull * BB * NBG * DI);
  u16*   vc    = (u16*)alloc(2ull * BB * NBG * DI);
  u16*   aO    = (u16*)alloc(2ull * BB * KK * DI);
  u16*   hmlp  = (u16*)alloc(2ull * BB * KK * MLPD);

  struct BW { u16 *sawq, *sawk, *sawv, *sawo, *cawq, *cawk, *cawv, *cawo, *w1, *w2; };
  BW bw[2];
  for (int blk = 0; blk < 2; ++blk) {
    bw[blk].sawq = (u16*)alloc(2ull * CC * DI);
    bw[blk].sawk = (u16*)alloc(2ull * CC * DI);
    bw[blk].sawv = (u16*)alloc(2ull * CC * DI);
    bw[blk].sawo = (u16*)alloc(2ull * DI * CC);
    bw[blk].cawq = (u16*)alloc(2ull * CC * DI);
    bw[blk].cawk = (u16*)alloc(2ull * CC * DI);
    bw[blk].cawv = (u16*)alloc(2ull * CC * DI);
    bw[blk].cawo = (u16*)alloc(2ull * DI * CC);
    bw[blk].w1   = (u16*)alloc(2ull * CC * MLPD);
    bw[blk].w2   = (u16*)alloc(2ull * MLPD * CC);
  }
  u16* convT = (u16*)alloc(2ull * DCC * CC);

  auto cvt = [&](const float* in, u16* out, size_t n) {
    f32_to_bf16_k<<<dim3((unsigned)(n / 256)), 256, 0, stream>>>(in, out);
  };
  auto gemm = [&](const u16* A, const u16* W, const float* bias, float* oF, u16* oH,
                  int M, int Kd, int Nn, int act, int addto) {
    gemm_bf16_k<<<dim3((unsigned)(Nn / 64), (unsigned)(M / 128)), 256, 0, stream>>>(
        A, W, bias, oF, oH, M, Kd, Nn, act, addto);
  };
  auto flash = [&](const u16* Qp, const u16* Kp, const u16* Vp, u16* Op, int NQ, int NK) {
    flash_attn_k<<<dim3((unsigned)(BB * NHH * (NQ / 64))), 128, 0, stream>>>(
        Qp, Kp, Vp, Op, NQ, NK, 0.125f);
  };

  // ---- weight conversion ----
  for (int blk = 0; blk < 2; ++blk) {
    int pb = blk ? 31 : 5;
    cvt(F(pb + 0),  bw[blk].sawq, (size_t)CC * DI);
    cvt(F(pb + 2),  bw[blk].sawk, (size_t)CC * DI);
    cvt(F(pb + 4),  bw[blk].sawv, (size_t)CC * DI);
    cvt(F(pb + 6),  bw[blk].sawo, (size_t)DI * CC);
    cvt(F(pb + 8),  bw[blk].cawq, (size_t)CC * DI);
    cvt(F(pb + 10), bw[blk].cawk, (size_t)CC * DI);
    cvt(F(pb + 12), bw[blk].cawv, (size_t)CC * DI);
    cvt(F(pb + 14), bw[blk].cawo, (size_t)DI * CC);
    cvt(F(pb + 16), bw[blk].w1,   (size_t)CC * MLPD);
    cvt(F(pb + 18), bw[blk].w2,   (size_t)MLPD * CC);
  }
  transpose_cw_k<<<dim3((DCC * CC) / 256), 256, 0, stream>>>(F(57), convT);

  // ---- gathers + depth path ----
  build_indices_k<<<BB, 32, 0, stream>>>(mask, d_idx, g_idx);
  gather_defect_k<<<dim3((unsigned)(((size_t)BB * KK * CC) / 256)), 256, 0, stream>>>(feat, d_idx, x1, x2);
  gather_bg_k<<<dim3((unsigned)(((size_t)BB * NBG * CC) / 256)), 256, 0, stream>>>(feat, g_idx, bg);
  depth_gather_k<<<dim3(BB * KK), 256, 0, stream>>>(depth, d_idx, ddin);
  gemm(ddin, convT, F(58), nullptr, ddef, BB * KK, DCC, CC, 0, 0); // 1x1 conv -> d_def

  // ---- two interact blocks (intra: ctx=background, inter: ctx=depth-defect) ----
  for (int blk = 0; blk < 2; ++blk) {
    int pb = blk ? 31 : 5;
    const float *sabq = F(pb + 1), *sabk = F(pb + 3), *sabv = F(pb + 5), *sabo = F(pb + 7);
    const float *cabq = F(pb + 9), *cabk = F(pb + 11), *cabv = F(pb + 13), *cabo = F(pb + 15);
    const float *b1 = F(pb + 17), *b2 = F(pb + 19);
    const float *n1g = F(pb + 20), *n1b = F(pb + 21);
    const float *n2g = F(pb + 22), *n2b = F(pb + 23);
    const float *n3g = F(pb + 24), *n3b = F(pb + 25);
    float* x = blk ? x2 : x1;
    const u16* ctx = blk ? ddef : bg;
    int NKctx = blk ? KK : NBG;
    BW& W = bw[blk];

    // cross-attn context projections (fixed ctx, ca weights)
    gemm(ctx, W.cawk, cabk, nullptr, kc, BB * NKctx, CC, DI, 0, 0);
    gemm(ctx, W.cawv, cabv, nullptr, vc, BB * NKctx, CC, DI, 0, 0);

    // self-attention
    layernorm_k<<<dim3(BB * KK), 256, 0, stream>>>(x, n1g, n1b, xln);
    gemm(xln, W.sawq, sabq, nullptr, qb,   BB * KK, CC, DI, 0, 0);
    gemm(xln, W.sawk, sabk, nullptr, kbuf, BB * KK, CC, DI, 0, 0);
    gemm(xln, W.sawv, sabv, nullptr, vbuf, BB * KK, CC, DI, 0, 0);
    flash(qb, kbuf, vbuf, aO, KK, KK);
    gemm(aO, W.sawo, sabo, x, nullptr, BB * KK, DI, CC, 0, 1);

    // MLP (exact-erf GELU)
    layernorm_k<<<dim3(BB * KK), 256, 0, stream>>>(x, n2g, n2b, xln);
    gemm(xln, W.w1, b1, nullptr, hmlp, BB * KK, CC, MLPD, 1, 0);
    gemm(hmlp, W.w2, b2, x, nullptr, BB * KK, MLPD, CC, 0, 1);

    // cross-attention
    layernorm_k<<<dim3(BB * KK), 256, 0, stream>>>(x, n3g, n3b, xln);
    gemm(xln, W.cawq, cabq, nullptr, qb, BB * KK, CC, DI, 0, 0);
    flash(qb, kc, vc, aO, KK, NKctx);
    gemm(aO, W.cawo, cabo, x, nullptr, BB * KK, DI, CC, 0, 1);
  }

  // ---- output: copy feat, then scatter new tokens ----
  (void)hipMemcpyAsync(d_out, feat, sizeof(float) * (size_t)out_size,
                       hipMemcpyDeviceToDevice, stream);
  scatter_out_k<<<dim3((unsigned)(((size_t)BB * KK * CC) / 256)), 256, 0, stream>>>(
      x1, x2, d_idx, (float*)d_out);
}
